// TextBlock_15470472200714
// MI455X (gfx1250) — compile-verified, matrix operand order checked by reading, not run on hardware
//
#include <hip/hip_runtime.h>
#include <hip/hip_bf16.h>

// ---------------------------------------------------------------------------
// Types for CDNA5 WMMA (wave32)
// ---------------------------------------------------------------------------
typedef __bf16 bf16_t;
typedef __attribute__((ext_vector_type(16))) __bf16 v16bf;
typedef __attribute__((ext_vector_type(8)))  __bf16 v8bf;
typedef __attribute__((ext_vector_type(8)))  float  v8f;
typedef int v4i __attribute__((vector_size(16)));

union AFrag {
    v16bf v;
    struct { v8bf lo, hi; } h;
};

// Fragment from LDS laid out [row][k], k contiguous. Per the CDNA5 ISA layout,
// lane L owns row/col L&15; half-wave selects K {kbase..+7} and {kbase+16..+23},
// kbase = 8*(L>>4). Two aligned ds_load_b128 per fragment.
__device__ __forceinline__ v16bf load_frag(const bf16_t* base, int row, int stride, int koff) {
    const bf16_t* p = base + row * stride + koff;
    AFrag f;
    f.h.lo = *(const v8bf*)(p);
    f.h.hi = *(const v8bf*)(p + 16);
    return f.v;
}

#define WMMA_BF16(A, B, C) \
    __builtin_amdgcn_wmma_f32_16x16x32_bf16(false, (A), false, (B), (short)0, (C), false, false)

// ---------------------------------------------------------------------------
// CDNA5 async global->LDS copy (16B per lane). Builtin params are generic
// pointers to int4 per the clang declaration; fallback keeps host pass legal.
// ---------------------------------------------------------------------------
#if __has_builtin(__builtin_amdgcn_global_load_async_to_lds_b128) && \
    __has_builtin(__builtin_amdgcn_s_wait_asynccnt)
#define USE_ASYNC_LDS 1
#else
#define USE_ASYNC_LDS 0
#endif

__device__ __forceinline__ void cp16(void* lds, const void* g) {
#if USE_ASYNC_LDS
    __builtin_amdgcn_global_load_async_to_lds_b128((v4i*)g, (v4i*)lds, 0, 0);
#else
    *(v8bf*)lds = *(const v8bf*)g;
#endif
}

__device__ __forceinline__ void async_wait() {
#if USE_ASYNC_LDS
    __builtin_amdgcn_s_wait_asynccnt(0);
#endif
}

// ---------------------------------------------------------------------------
// Problem constants
// ---------------------------------------------------------------------------
#define BATCH 8
#define DIM   256
#define NTOK  1024
#define NHEAD 16
#define KDIM  16
#define DHEAD 64
#define DH    1024
#define HID   1024

// ---------------------------------------------------------------------------
// float -> bf16 (weights)
// ---------------------------------------------------------------------------
__global__ void cvt_bf16_kernel(const float* __restrict__ src, bf16_t* __restrict__ dst, int n) {
    int i = blockIdx.x * blockDim.x + threadIdx.x;
    if (i < n) dst[i] = (bf16_t)src[i];
}

// ---------------------------------------------------------------------------
// BN stats over (B,N): fold into y = x*s + t (biased var, eps=1e-5)
// ---------------------------------------------------------------------------
__global__ __launch_bounds__(256)
void bn_stats_kernel(const float* __restrict__ x, const float* __restrict__ g,
                     const float* __restrict__ bet, float* __restrict__ s,
                     float* __restrict__ t) {
    __shared__ float ssum[256];
    __shared__ float ssq[256];
    const int c = blockIdx.x;
    float sum = 0.f, sq = 0.f;
    for (int b = 0; b < BATCH; ++b) {
        const float* p = x + ((size_t)b * DIM + c) * NTOK;
        for (int n = threadIdx.x; n < NTOK; n += 256) {
            float v = p[n];
            sum += v; sq += v * v;
        }
    }
    ssum[threadIdx.x] = sum; ssq[threadIdx.x] = sq;
    __syncthreads();
    for (int off = 128; off > 0; off >>= 1) {
        if (threadIdx.x < off) {
            ssum[threadIdx.x] += ssum[threadIdx.x + off];
            ssq[threadIdx.x]  += ssq[threadIdx.x + off];
        }
        __syncthreads();
    }
    if (threadIdx.x == 0) {
        const float cnt = (float)BATCH * (float)NTOK;
        float m   = ssum[0] / cnt;
        float var = ssq[0] / cnt - m * m;
        float sc  = g[c] * rsqrtf(var + 1e-5f);
        s[c] = sc;
        t[c] = bet[c] - m * sc;
    }
}

// ---------------------------------------------------------------------------
// Fused BN-apply + transpose: x[b][c][n] (f32) -> yt[b][n][c] (bf16)
// ---------------------------------------------------------------------------
__global__ __launch_bounds__(256)
void bn_apply_t_kernel(const float* __restrict__ x, const float* __restrict__ s,
                       const float* __restrict__ t, bf16_t* __restrict__ yt) {
    __shared__ bf16_t tile[32][33];
    const int b  = blockIdx.z;
    const int c0 = blockIdx.y * 32;
    const int n0 = blockIdx.x * 32;
    const int tx = threadIdx.x & 31, ty = threadIdx.x >> 5;
    for (int rr = 0; rr < 32; rr += 8) {
        int c = c0 + ty + rr;
        tile[ty + rr][tx] = (bf16_t)(x[((size_t)b * DIM + c) * NTOK + n0 + tx] * s[c] + t[c]);
    }
    __syncthreads();
    for (int rr = 0; rr < 32; rr += 8)
        yt[((size_t)b * NTOK + n0 + ty + rr) * DIM + c0 + tx] = tile[tx][ty + rr];
}

// ---------------------------------------------------------------------------
// bf16 batched transpose: in[b][r][c] -> out[b][c][r]
// ---------------------------------------------------------------------------
__global__ __launch_bounds__(256)
void transpose_bf16_kernel(const bf16_t* __restrict__ in, bf16_t* __restrict__ out,
                           int R, int C) {
    __shared__ bf16_t tile[32][33];
    const int b  = blockIdx.z;
    const int r0 = blockIdx.y * 32;
    const int c0 = blockIdx.x * 32;
    const int tx = threadIdx.x & 31, ty = threadIdx.x >> 5;
    const bf16_t* inb = in + (size_t)b * R * C;
    bf16_t* outb = out + (size_t)b * R * C;
    for (int rr = 0; rr < 32; rr += 8)
        tile[ty + rr][tx] = inb[(size_t)(r0 + ty + rr) * C + c0 + tx];
    __syncthreads();
    for (int rr = 0; rr < 32; rr += 8)
        outb[(size_t)(c0 + ty + rr) * R + r0 + tx] = tile[tx][ty + rr];
}

// ---------------------------------------------------------------------------
// Tiled bf16 WMMA GEMM: Out[b][o][n] = W[o][k] @ Yt[b][n][k]^T + bias (+res)
//   workgroup tile 64(O) x 128(N); 8 waves 2x4; wave tile 32x32; K step 32
//   All LDS staging via 16B async copies.
//   mode 0: outb = bf16(acc+bias)   mode 1: outf = acc+bias+res (f32)
//   mode 2: outb = bf16(relu6(acc+bias))
// ---------------------------------------------------------------------------
#define AS_STRIDE 40   // 80B rows: 16B-aligned segments, conflict-free frags
#define BS_STRIDE 40

__global__ __launch_bounds__(256)
void gemm_kernel(const bf16_t* __restrict__ W, const bf16_t* __restrict__ Yt,
                 const float* __restrict__ bias, const float* __restrict__ res,
                 float* __restrict__ outf, bf16_t* __restrict__ outb,
                 int O, int K, int N, int mode) {
    __shared__ bf16_t As[64 * AS_STRIDE];
    __shared__ bf16_t Bs[128 * BS_STRIDE];

    const int tid  = threadIdx.x;
    const int lane = tid & 31;
    const int wave = tid >> 5;
    const int half = lane >> 4;
    const int col  = lane & 15;

    const int b  = blockIdx.z;
    const int o0 = blockIdx.y * 64;
    const int n0 = blockIdx.x * 128;
    const bf16_t* Yb = Yt + (size_t)b * N * K;

    const int ow = (wave & 1) * 32;
    const int nw = (wave >> 1) * 32;

    const int ar = tid >> 2, asg = tid & 3;         // A: 64 rows x 4 segs
    const int br0 = tid >> 1, bs0 = (tid & 1) * 2;  // B: 128 rows x 4 segs (2/thread)

    v8f acc[2][2] = {};

    for (int kk = 0; kk < K; kk += 32) {
        cp16(&As[ar * AS_STRIDE + asg * 8], W + (size_t)(o0 + ar) * K + kk + asg * 8);
        cp16(&Bs[br0 * BS_STRIDE + bs0 * 8],       Yb + (size_t)(n0 + br0) * K + kk + bs0 * 8);
        cp16(&Bs[br0 * BS_STRIDE + (bs0 + 1) * 8], Yb + (size_t)(n0 + br0) * K + kk + (bs0 + 1) * 8);
        if (kk + 32 < K)
            __builtin_prefetch(W + (size_t)(o0 + ar) * K + kk + 32, 0, 1);
        async_wait();
        __syncthreads();

        v16bf a0 = load_frag(As, ow + col,      AS_STRIDE, half * 8);
        v16bf a1 = load_frag(As, ow + 16 + col, AS_STRIDE, half * 8);
        v16bf b0 = load_frag(Bs, nw + col,      BS_STRIDE, half * 8);
        v16bf b1 = load_frag(Bs, nw + 16 + col, BS_STRIDE, half * 8);

        acc[0][0] = WMMA_BF16(a0, b0, acc[0][0]);
        acc[0][1] = WMMA_BF16(a0, b1, acc[0][1]);
        acc[1][0] = WMMA_BF16(a1, b0, acc[1][0]);
        acc[1][1] = WMMA_BF16(a1, b1, acc[1][1]);
        __syncthreads();
    }

    // epilogue: acc element e -> (M = e+8*half, N = lane&15); stores coalesce along n
    for (int mi = 0; mi < 2; ++mi) {
        for (int ni = 0; ni < 2; ++ni) {
            #pragma unroll
            for (int e = 0; e < 8; ++e) {
                int om = o0 + ow + mi * 16 + e + 8 * half;
                int on = n0 + nw + ni * 16 + col;
                float v = acc[mi][ni][e] + bias[om];
                size_t idx = ((size_t)b * O + om) * N + on;
                if (mode == 0)      outb[idx] = (bf16_t)v;
                else if (mode == 1) outf[idx] = v + res[idx];
                else                outb[idx] = (bf16_t)fminf(fmaxf(v, 0.f), 6.f);
            }
        }
    }
}

// ---------------------------------------------------------------------------
// Flash attention, one workgroup per (b, head, 128-query block).
// qt/kt: [b][n][c] bf16 (c = h*16+kd, contiguous)  v: [b][h*64+d][n] bf16
// out:   [b][n][dh] bf16 (k-contiguous for the projection GEMM)
// scores *= sqrt(kd) = 4 per the reference. kd padded 16->32.
// ---------------------------------------------------------------------------
#define QS_STRIDE 40
#define KS_STRIDE 40
#define VS_STRIDE 136
#define SS_STRIDE 132
#define PS_STRIDE 136
#define ATTN_LDS_BYTES (128*QS_STRIDE*2 + 128*KS_STRIDE*2 + 64*VS_STRIDE*2 + \
                        8*16*SS_STRIDE*4 + 8*16*PS_STRIDE*2 + 8*16*4)

__global__ __launch_bounds__(256)
void attn_kernel(const bf16_t* __restrict__ qt, const bf16_t* __restrict__ kt,
                 const bf16_t* __restrict__ vb, bf16_t* __restrict__ ot) {
    extern __shared__ char smem[];
    bf16_t* Qs = (bf16_t*)smem;                       // [128][QS_STRIDE]
    bf16_t* Ks = Qs + 128 * QS_STRIDE;                // [128][KS_STRIDE]
    bf16_t* Vs = Ks + 128 * KS_STRIDE;                // [64][VS_STRIDE]
    float*  Ss = (float*)(Vs + 64 * VS_STRIDE);       // per-wave [16][SS_STRIDE]
    bf16_t* Ps = (bf16_t*)(Ss + 8 * 16 * SS_STRIDE);  // per-wave [16][PS_STRIDE]
    float*  aux = (float*)(Ps + 8 * 16 * PS_STRIDE);  // per-wave [16]

    const int tid  = threadIdx.x;
    const int lane = tid & 31;
    const int wave = tid >> 5;
    const int half = lane >> 4;
    const int col  = lane & 15;

    const int q0 = blockIdx.x * 128;
    const int h  = blockIdx.y;
    const int b  = blockIdx.z;
    const size_t headv = ((size_t)b * DH + (size_t)h * DHEAD) * NTOK;

    // zero kd-pad [16..32) of Qs and Ks (async copies only touch [0..16))
    {
        v8bf z = {};
        int row = tid >> 1, hh = tid & 1;
        *(v8bf*)(&Qs[row * QS_STRIDE + 16 + hh * 8]) = z;
        *(v8bf*)(&Ks[row * KS_STRIDE + 16 + hh * 8]) = z;
    }
    // stage Q block: Qs[n][kd] <- qt[b][q0+n][h*16 + kd], 2x16B per row
    {
        int n = tid >> 1, sg = tid & 1;
        cp16(&Qs[n * QS_STRIDE + sg * 8],
             qt + ((size_t)b * NTOK + q0 + n) * DIM + h * KDIM + sg * 8);
    }
    async_wait();
    __syncthreads();

    const v16bf qf = load_frag(Qs, wave * 16 + col, QS_STRIDE, half * 8);

    v8f acc[4] = {};
    float m = -1e30f, l = 0.f;        // kept consistent in both half-waves
    float*  myS   = Ss  + wave * 16 * SS_STRIDE;
    bf16_t* myP   = Ps  + wave * 16 * PS_STRIDE;
    float*  myAux = aux + wave * 16;

    for (int c0 = 0; c0 < NTOK; c0 += 128) {
        // stage K chunk: Ks[key][kd], 2x16B per row
        {
            int n = tid >> 1, sg = tid & 1;
            cp16(&Ks[n * KS_STRIDE + sg * 8],
                 kt + ((size_t)b * NTOK + c0 + n) * DIM + h * KDIM + sg * 8);
        }
        // stage V chunk: Vs[d][key], 16x16B per row (4 segs/thread)
        #pragma unroll
        for (int r = 0; r < 4; ++r) {
            int s = tid + 256 * r;
            int d = s >> 4, sg = s & 15;
            cp16(&Vs[d * VS_STRIDE + sg * 8], vb + headv + (size_t)d * NTOK + c0 + sg * 8);
        }
        async_wait();
        __syncthreads();

        // S = (Q @ K^T) * 4 : 8 key sub-tiles
        #pragma unroll
        for (int jt = 0; jt < 8; ++jt) {
            v16bf kf = load_frag(Ks, jt * 16 + col, KS_STRIDE, half * 8);
            v8f s = {};
            s = WMMA_BF16(qf, kf, s);
            #pragma unroll
            for (int e = 0; e < 8; ++e)
                myS[(e + 8 * half) * SS_STRIDE + jt * 16 + col] = s[e] * 4.0f;
        }

        // online softmax: lane handles row col, half-wave splits 128 keys in two
        {
            float* srow  = myS + col * SS_STRIDE + half * 64;
            bf16_t* prow = myP + col * PS_STRIDE + half * 64;
            float cm = -1e30f;
            for (int j = 0; j < 64; ++j) cm = fmaxf(cm, srow[j]);
            cm = fmaxf(cm, __shfl_xor(cm, 16, 32));
            float nm    = fmaxf(m, cm);
            float alpha = __expf(m - nm);
            float psum  = 0.f;
            for (int j = 0; j < 64; ++j) {
                float p = __expf(srow[j] - nm);
                psum += p;
                prow[j] = (bf16_t)p;
            }
            psum += __shfl_xor(psum, 16, 32);
            l = l * alpha + psum;
            m = nm;
            if (lane < 16) myAux[col] = alpha;
        }

        // rescale accumulators by alpha(row); DS ops are wave-ordered
        #pragma unroll
        for (int e = 0; e < 8; ++e) {
            float al = myAux[e + 8 * half];
            #pragma unroll
            for (int dt = 0; dt < 4; ++dt) acc[dt][e] *= al;
        }

        // O += P @ V
        #pragma unroll
        for (int kt4 = 0; kt4 < 4; ++kt4) {
            v16bf pf = load_frag(myP, col, PS_STRIDE, kt4 * 32 + half * 8);
            #pragma unroll
            for (int dt = 0; dt < 4; ++dt) {
                v16bf vf = load_frag(Vs, dt * 16 + col, VS_STRIDE, kt4 * 32 + half * 8);
                acc[dt] = WMMA_BF16(pf, vf, acc[dt]);
            }
        }
        __syncthreads();
    }

    // epilogue: ot[b][q][h*64+d] — lanes vary d, contiguous stores
    if (lane < 16) myAux[col] = 1.0f / l;
    #pragma unroll
    for (int e = 0; e < 8; ++e) {
        float li = myAux[e + 8 * half];
        int qg = q0 + wave * 16 + e + 8 * half;
        #pragma unroll
        for (int dt = 0; dt < 4; ++dt) {
            int dg = dt * 16 + col;
            ot[((size_t)b * NTOK + qg) * DH + h * DHEAD + dg] = (bf16_t)(acc[dt][e] * li);
        }
    }
}

// ---------------------------------------------------------------------------
// Host orchestration
// ---------------------------------------------------------------------------
extern "C" void kernel_launch(void* const* d_in, const int* in_sizes, int n_in,
                              void* d_out, int out_size, void* d_ws, size_t ws_size,
                              hipStream_t stream) {
    const float* x   = (const float*)d_in[0];
    const float* g1  = (const float*)d_in[1];
    const float* b1  = (const float*)d_in[2];
    const float* wq  = (const float*)d_in[3];
    const float* bq  = (const float*)d_in[4];
    const float* wk  = (const float*)d_in[5];
    const float* bk  = (const float*)d_in[6];
    const float* wv  = (const float*)d_in[7];
    const float* bv  = (const float*)d_in[8];
    const float* wp  = (const float*)d_in[9];
    const float* bp  = (const float*)d_in[10];
    const float* g2  = (const float*)d_in[11];
    const float* b2  = (const float*)d_in[12];
    const float* w1  = (const float*)d_in[13];
    const float* bb1 = (const float*)d_in[14];
    const float* w2  = (const float*)d_in[15];
    const float* bb2 = (const float*)d_in[16];
    float* out = (float*)d_out;

    char* ws = (char*)d_ws;
    size_t off = 0;
    auto alloc = [&](size_t bytes) -> void* {
        void* p = ws + off;
        off = (off + bytes + 255) & ~(size_t)255;
        return p;
    };

    const size_t BCN = (size_t)BATCH * DIM * NTOK;
    const size_t BHN = (size_t)BATCH * DH * NTOK;

    float*  s1   = (float*)alloc(DIM * 4);
    float*  t1   = (float*)alloc(DIM * 4);
    float*  s2   = (float*)alloc(DIM * 4);
    float*  t2   = (float*)alloc(DIM * 4);
    bf16_t* wqb  = (bf16_t*)alloc((size_t)DIM * DIM * 2);
    bf16_t* wkb  = (bf16_t*)alloc((size_t)DIM * DIM * 2);
    bf16_t* wvb  = (bf16_t*)alloc((size_t)DH * DIM * 2);
    bf16_t* wpb  = (bf16_t*)alloc((size_t)DIM * DH * 2);
    bf16_t* w1b  = (bf16_t*)alloc((size_t)HID * DIM * 2);
    bf16_t* w2b  = (bf16_t*)alloc((size_t)DIM * HID * 2);
    bf16_t* ytb  = (bf16_t*)alloc(BCN * 2);   // BN1(x), [b][n][c]
    bf16_t* qn   = (bf16_t*)alloc(BCN * 2);   // Q, [b][o][n]
    bf16_t* kn   = (bf16_t*)alloc(BCN * 2);   // K, [b][o][n]
    bf16_t* qtb  = (bf16_t*)alloc(BCN * 2);   // Q, [b][n][o]
    bf16_t* ktb  = (bf16_t*)alloc(BCN * 2);   // K, [b][n][o]
    bf16_t* vbf  = (bf16_t*)alloc(BHN * 2);   // V, [b][dh][n]
    bf16_t* otb  = (bf16_t*)alloc(BHN * 2);   // attn out, [b][n][dh]
    float*  xf2  = (float*)alloc(BCN * 4);    // residual 1, [b][c][n]
    bf16_t* y2tb = (bf16_t*)alloc(BCN * 2);   // BN2(xf2), [b][n][c]
    bf16_t* hn   = (bf16_t*)alloc(BHN * 2);   // MLP hidden, [b][h][n]
    bf16_t* htb  = (bf16_t*)alloc(BHN * 2);   // MLP hidden, [b][n][h]

    // weights -> bf16
    cvt_bf16_kernel<<<(DIM * DIM + 255) / 256, 256, 0, stream>>>(wq, wqb, DIM * DIM);
    cvt_bf16_kernel<<<(DIM * DIM + 255) / 256, 256, 0, stream>>>(wk, wkb, DIM * DIM);
    cvt_bf16_kernel<<<(DH * DIM + 255) / 256, 256, 0, stream>>>(wv, wvb, DH * DIM);
    cvt_bf16_kernel<<<(DIM * DH + 255) / 256, 256, 0, stream>>>(wp, wpb, DIM * DH);
    cvt_bf16_kernel<<<(HID * DIM + 255) / 256, 256, 0, stream>>>(w1, w1b, HID * DIM);
    cvt_bf16_kernel<<<(DIM * HID + 255) / 256, 256, 0, stream>>>(w2, w2b, DIM * HID);

    // BN1 folded + transpose: ytb[b][n][c]
    bn_stats_kernel<<<DIM, 256, 0, stream>>>(x, g1, b1, s1, t1);
    bn_apply_t_kernel<<<dim3(NTOK / 32, DIM / 32, BATCH), 256, 0, stream>>>(x, s1, t1, ytb);

    // Q, K, V projections
    gemm_kernel<<<dim3(NTOK / 128, DIM / 64, BATCH), 256, 0, stream>>>(
        wqb, ytb, bq, nullptr, nullptr, qn, DIM, DIM, NTOK, 0);
    gemm_kernel<<<dim3(NTOK / 128, DIM / 64, BATCH), 256, 0, stream>>>(
        wkb, ytb, bk, nullptr, nullptr, kn, DIM, DIM, NTOK, 0);
    gemm_kernel<<<dim3(NTOK / 128, DH / 64, BATCH), 256, 0, stream>>>(
        wvb, ytb, bv, nullptr, nullptr, vbf, DH, DIM, NTOK, 0);

    // Q,K -> token-major for the attention A/B fragments
    transpose_bf16_kernel<<<dim3(NTOK / 32, DIM / 32, BATCH), 256, 0, stream>>>(qn, qtb, DIM, NTOK);
    transpose_bf16_kernel<<<dim3(NTOK / 32, DIM / 32, BATCH), 256, 0, stream>>>(kn, ktb, DIM, NTOK);

    // attention -> otb[b][n][dh]
    attn_kernel<<<dim3(NTOK / 128, NHEAD, BATCH), 256, ATTN_LDS_BYTES, stream>>>(
        qtb, ktb, vbf, otb);

    // projection + residual: xf2 = x + Wp @ o + bp
    gemm_kernel<<<dim3(NTOK / 128, DIM / 64, BATCH), 256, 0, stream>>>(
        wpb, otb, bp, x, xf2, nullptr, DIM, DH, NTOK, 1);

    // BN2 folded + transpose
    bn_stats_kernel<<<DIM, 256, 0, stream>>>(xf2, g2, b2, s2, t2);
    bn_apply_t_kernel<<<dim3(NTOK / 32, DIM / 32, BATCH), 256, 0, stream>>>(xf2, s2, t2, y2tb);

    // MLP: h = relu6(W1 @ y2 + bb1); out = xf2 + W2 @ h + bb2
    gemm_kernel<<<dim3(NTOK / 128, HID / 64, BATCH), 256, 0, stream>>>(
        w1b, y2tb, bb1, nullptr, nullptr, hn, HID, DIM, NTOK, 2);
    transpose_bf16_kernel<<<dim3(NTOK / 32, HID / 32, BATCH), 256, 0, stream>>>(hn, htb, HID, NTOK);
    gemm_kernel<<<dim3(NTOK / 128, DIM / 64, BATCH), 256, 0, stream>>>(
        w2b, htb, bb2, xf2, out, nullptr, DIM, HID, NTOK, 1);
}
